// GraphEncoder_40664750359187
// MI455X (gfx1250) — matrix operand order repaired and checked by reading, not compiled, not run on hardware
//
#include <hip/hip_runtime.h>
#include <hip/hip_bf16.h>
#include <math.h>

// ---------------------------------------------------------------------------
// GraphEncoder (3-layer GAT + pooling + MLP head) for MI455X / gfx1250.
// Dense GEMMs run on v_wmma_f32_16x16x32_f16 (wave32 WMMA), edge softmax via
// native float atomics, pooling fused into epilogues.
// ---------------------------------------------------------------------------

typedef __attribute__((ext_vector_type(16))) _Float16 v16h;
typedef __attribute__((ext_vector_type(8)))  _Float16 v8h;
typedef __attribute__((ext_vector_type(8)))  float    v8f;

union HFrag { v16h v; v8h h[2]; };

#define TPB 256
#define HC  256          // H*C
#define GCOLS 832        // F*(1+3H)
#define LATD 256
#define NEG_SLOPE 0.2f

// ---------------------------------------------------------------- fill
__global__ void k_fill(float* __restrict__ p, float v, long n) {
    long t = (long)blockIdx.x * blockDim.x + threadIdx.x;
    if (t < n) p[t] = v;
}

// ------------------------------------------------- pack W [K,Ncols] f32 ->
// B-fragment order: Bp[((ntile*ksteps+ks)*32 + lane)*16 + i] = W[k][n]
// with k = ks*32 + i + 16*(lane>>4), n = ntile*16 + (lane&15).
__global__ void k_pack_b(const float* __restrict__ W, _Float16* __restrict__ Bp,
                         int K, int Ncols) {
    int t = blockIdx.x * blockDim.x + threadIdx.x;
    int total = K * Ncols;
    if (t >= total) return;
    int i     = t & 15;
    int lane  = (t >> 4) & 31;
    int rest  = t >> 9;               // ntile*ksteps + ks
    int ksteps = K >> 5;
    int ks    = rest % ksteps;
    int ntile = rest / ksteps;
    int krow  = ks * 32 + i + ((lane >> 4) << 4);
    int col   = ntile * 16 + (lane & 15);
    Bp[t] = (_Float16)W[(size_t)krow * Ncols + col];
}

// ------------------------------------------------- input MLP: n0 = relu(bf@W+b)
// K=5 -> plain VALU; fuse pooling atomicMax and f16 recast.
__global__ void k_in_mlp(const float* __restrict__ bf, const float* __restrict__ Wf,
                         const float* __restrict__ bb, const int* __restrict__ batch,
                         _Float16* __restrict__ x0h, float* __restrict__ g, int N) {
    int t = blockIdx.x * blockDim.x + threadIdx.x;
    if (t >= N * 64) return;
    int n = t >> 6, f = t & 63;
    float acc = bb[f];
#pragma unroll
    for (int k = 0; k < 5; ++k) acc += bf[n * 5 + k] * Wf[k * 64 + f];
    acc = fmaxf(acc, 0.0f);
    x0h[t] = (_Float16)acc;
    atomicMax(&g[(size_t)batch[n] * GCOLS + f], acc);
}

// ------------------------------------------------- WMMA GEMM  C = A(f16) @ Bp
// A row-major [M,K] f16; Bp pre-packed fragments; one wave = one 16x16 tile.
__global__ void k_wmma_gemm(const _Float16* __restrict__ A,
                            const _Float16* __restrict__ Bp,
                            const float* __restrict__ bias,
                            float* __restrict__ Cf, _Float16* __restrict__ Ch,
                            int M, int K, int Ncols, int applyRelu) {
    const int lane  = threadIdx.x & 31;
    const int wave  = threadIdx.x >> 5;
    const int mtile = blockIdx.x;
    const int ntile = blockIdx.y * 8 + wave;
    const int ksteps = K >> 5;
    const int hi = lane >> 4;                         // 0 or 1
    int row = mtile * 16 + (lane & 15);
    int rowL = row < M ? row : (M - 1);               // clamp for ragged M
    const _Float16* arow = A + (size_t)rowL * K;
    const _Float16* bptr = Bp + ((size_t)ntile * ksteps) * 512 + lane * 16;

    v8f acc = {};
    for (int ks = 0; ks < ksteps; ++ks) {
        const int kk = ks << 5;
        HFrag a, b;
        a.h[0] = *(const v8h*)(arow + kk + hi * 16);       // K = kk+16*hi .. +7
        a.h[1] = *(const v8h*)(arow + kk + hi * 16 + 8);   // K = kk+16*hi+8 ..
        b.v    = *(const v16h*)(bptr);
        bptr  += 512;
        acc = __builtin_amdgcn_wmma_f32_16x16x32_f16(
            false, a.v, false, b.v, (short)0, acc, false, false);
    }

    const int col = ntile * 16 + (lane & 15);
    const float bv = bias ? bias[col] : 0.0f;
#pragma unroll
    for (int j = 0; j < 8; ++j) {
        int r = mtile * 16 + j + hi * 8;                   // C/D VGPR layout
        if (r >= M) continue;
        float v = acc[j] + bv;
        if (applyRelu) v = fmaxf(v, 0.0f);
        size_t idx = (size_t)r * Ncols + col;
        if (Cf) Cf[idx] = v;
        if (Ch) Ch[idx] = (_Float16)v;
    }
}

// ------------------------------------------------- per-node attention coeffs
// one wave per node; lane covers 8 contiguous channels; head = lane>>3.
__global__ void k_att(const float* __restrict__ h, const float* __restrict__ asrc,
                      const float* __restrict__ adst, float* __restrict__ a_s,
                      float* __restrict__ a_d, int N) {
    int gw = (blockIdx.x * blockDim.x + threadIdx.x) >> 5;
    int lane = threadIdx.x & 31;
    if (gw >= N) return;
    const float* hr = h + (size_t)gw * HC + lane * 8;
    int head = lane >> 3;
    int cb = (lane & 7) * 8;
    float ss = 0.f, sd = 0.f;
#pragma unroll
    for (int j = 0; j < 8; ++j) {
        float v = hr[j];
        ss += v * asrc[head * 64 + cb + j];
        sd += v * adst[head * 64 + cb + j];
    }
#pragma unroll
    for (int m = 1; m < 8; m <<= 1) {       // reduce within 8-lane head group
        ss += __shfl_xor(ss, m, 32);
        sd += __shfl_xor(sd, m, 32);
    }
    if ((lane & 7) == 0) {
        a_s[gw * 4 + head] = ss;
        a_d[gw * 4 + head] = sd;
    }
}

// ------------------------------------------------- edge pass 1: leaky-relu + max
__global__ void k_edge_max(const int* __restrict__ ei, const float* __restrict__ a_s,
                           const float* __restrict__ a_d, float* __restrict__ ebuf,
                           float* __restrict__ mmax, int E, int N) {
    int t = blockIdx.x * blockDim.x + threadIdx.x;
    int Etot = E + N;
    if (t >= Etot * 4) return;
    int e = t >> 2, hd = t & 3, s, d;
    if (e < E) { s = ei[e]; d = ei[E + e]; } else { s = d = e - E; }
    float v = a_s[s * 4 + hd] + a_d[d * 4 + hd];
    v = v > 0.f ? v : NEG_SLOPE * v;
    ebuf[t] = v;
    atomicMax(&mmax[d * 4 + hd], v);
}

// ------------------------------------------------- edge pass 2: exp + sum
__global__ void k_edge_exp(const int* __restrict__ ei, float* __restrict__ ebuf,
                           const float* __restrict__ mmax, float* __restrict__ ssum,
                           int E, int N) {
    int t = blockIdx.x * blockDim.x + threadIdx.x;
    int Etot = E + N;
    if (t >= Etot * 4) return;
    int e = t >> 2, hd = t & 3, d;
    if (e < E) { d = ei[E + e]; } else { d = e - E; }
    float p = __expf(ebuf[t] - mmax[d * 4 + hd]);
    ebuf[t] = p;
    atomicAdd(&ssum[d * 4 + hd], p);
}

// ------------------------------------------------- edge pass 3: scatter alpha*h
// one wave per edge; lane handles 8 contiguous channels of its head.
__global__ void k_edge_scatter(const int* __restrict__ ei, const float* __restrict__ ebuf,
                               const float* __restrict__ ssum, const float* __restrict__ h,
                               float* __restrict__ out, int E, int N) {
    int e = (blockIdx.x * blockDim.x + threadIdx.x) >> 5;
    int lane = threadIdx.x & 31;
    int Etot = E + N;
    if (e >= Etot) return;
    int s, d;
    if (e < E) { s = ei[e]; d = ei[E + e]; } else { s = d = e - E; }
    int hd = lane >> 3;
    float alpha = ebuf[e * 4 + hd] / ssum[d * 4 + hd];
    const float* hr = h + (size_t)s * HC + lane * 8;
    float* orow = out + (size_t)d * HC + lane * 8;
#pragma unroll
    for (int j = 0; j < 8; ++j) atomicAdd(&orow[j], alpha * hr[j]);
}

// ------------------------------------------------- layer epilogue: +bias,relu,
// f16 recast for next GEMM, fused pooling atomicMax.
__global__ void k_epilogue(const float* __restrict__ out, const float* __restrict__ bias,
                           const int* __restrict__ batch, _Float16* __restrict__ xh,
                           float* __restrict__ g, int N, int poolOff) {
    int t = blockIdx.x * blockDim.x + threadIdx.x;
    if (t >= N * HC) return;
    int n = t >> 8, c = t & 255;
    float v = fmaxf(out[t] + bias[c], 0.0f);
    xh[t] = (_Float16)v;
    atomicMax(&g[(size_t)batch[n] * GCOLS + poolOff + c], v);
}

// ------------------------------------------------- f32 -> f16
__global__ void k_f2h(const float* __restrict__ s, _Float16* __restrict__ d, int n) {
    int t = blockIdx.x * blockDim.x + threadIdx.x;
    if (t < n) d[t] = (_Float16)s[t];
}

// ===========================================================================
extern "C" void kernel_launch(void* const* d_in, const int* in_sizes, int n_in,
                              void* d_out, int out_size, void* d_ws, size_t ws_size,
                              hipStream_t stream) {
    const float* bf    = (const float*)d_in[0];
    const int*   ei    = (const int*)  d_in[1];
    const int*   batch = (const int*)  d_in[2];
    const float* bfW   = (const float*)d_in[3];
    const float* bfb   = (const float*)d_in[4];
    const float* Wl[3]  = {(const float*)d_in[5],  (const float*)d_in[9],  (const float*)d_in[13]};
    const float* asl[3] = {(const float*)d_in[6],  (const float*)d_in[10], (const float*)d_in[14]};
    const float* adl[3] = {(const float*)d_in[7],  (const float*)d_in[11], (const float*)d_in[15]};
    const float* bl[3]  = {(const float*)d_in[8],  (const float*)d_in[12], (const float*)d_in[16]};
    const float* aggW  = (const float*)d_in[17];
    const float* aggb  = (const float*)d_in[18];
    const float* muW   = (const float*)d_in[19];
    const float* mub   = (const float*)d_in[20];
    const float* varW  = (const float*)d_in[21];
    const float* varb  = (const float*)d_in[22];
    float* outp = (float*)d_out;

    const int N = in_sizes[0] / 5;
    const int E = in_sizes[1] / 2;
    const int G = out_size / (2 * LATD);
    const int Etot = E + N;

    // ---------------- workspace carve (256B aligned slices) ----------------
    size_t off = 0;
    auto carve = [&](size_t bytes) -> void* {
        void* p = (char*)d_ws + off;
        off += (bytes + 255) & ~(size_t)255;
        return p;
    };
    float*    hbuf  = (float*)   carve((size_t)N * HC * 4);     // GEMM output h
    float*    obuf  = (float*)   carve((size_t)N * HC * 4);     // aggregation
    _Float16* xh    = (_Float16*)carve((size_t)N * HC * 2);     // layer input f16
    _Float16* x0h   = (_Float16*)carve((size_t)N * 64 * 2);     // layer-1 input
    float*    a_s   = (float*)   carve((size_t)N * 4 * 4);
    float*    a_d   = (float*)   carve((size_t)N * 4 * 4);
    float*    mmax  = (float*)   carve((size_t)N * 4 * 4);
    float*    ssum  = (float*)   carve((size_t)N * 4 * 4);
    float*    ebuf  = (float*)   carve((size_t)Etot * 4 * 4);
    float*    g     = (float*)   carve((size_t)G * GCOLS * 4);
    _Float16* g16   = (_Float16*)carve((size_t)G * GCOLS * 2);
    _Float16* lat16 = (_Float16*)carve((size_t)G * LATD * 2);
    _Float16* Wp[3];
    Wp[0] = (_Float16*)carve((size_t)64  * HC * 2);
    Wp[1] = (_Float16*)carve((size_t)HC  * HC * 2);
    Wp[2] = (_Float16*)carve((size_t)HC  * HC * 2);
    _Float16* aggp = (_Float16*)carve((size_t)GCOLS * LATD * 2);
    _Float16* mup  = (_Float16*)carve((size_t)LATD * LATD * 2);
    _Float16* varp = (_Float16*)carve((size_t)LATD * LATD * 2);
    (void)ws_size; (void)n_in;

    auto blocks = [](long n) { return (unsigned)((n + TPB - 1) / TPB); };

    // ---------------- weight packing (f32 -> WMMA B-fragment f16) ----------
    k_pack_b<<<blocks(64  * HC), TPB, 0, stream>>>(Wl[0], Wp[0], 64,  HC);
    k_pack_b<<<blocks(HC  * HC), TPB, 0, stream>>>(Wl[1], Wp[1], HC,  HC);
    k_pack_b<<<blocks(HC  * HC), TPB, 0, stream>>>(Wl[2], Wp[2], HC,  HC);
    k_pack_b<<<blocks(GCOLS*LATD),TPB,0, stream>>>(aggW,  aggp, GCOLS, LATD);
    k_pack_b<<<blocks(LATD*LATD), TPB,0, stream>>>(muW,   mup,  LATD, LATD);
    k_pack_b<<<blocks(LATD*LATD), TPB,0, stream>>>(varW,  varp, LATD, LATD);

    // ---------------- pooled buffer init + input MLP -----------------------
    k_fill<<<blocks((long)G * GCOLS), TPB, 0, stream>>>(g, -1e30f, (long)G * GCOLS);
    k_in_mlp<<<blocks((long)N * 64), TPB, 0, stream>>>(bf, bfW, bfb, batch, x0h, g, N);

    // ---------------- 3 GAT layers -----------------------------------------
    const int Kin[3]     = {64, HC, HC};
    const int poolOff[3] = {64, 64 + HC, 64 + 2 * HC};
    for (int l = 0; l < 3; ++l) {
        const _Float16* xin = (l == 0) ? x0h : xh;
        k_fill<<<blocks((long)N * 4),  TPB, 0, stream>>>(mmax, -1e30f, (long)N * 4);
        k_fill<<<blocks((long)N * 4),  TPB, 0, stream>>>(ssum, 0.0f,   (long)N * 4);
        k_fill<<<blocks((long)N * HC), TPB, 0, stream>>>(obuf, 0.0f,   (long)N * HC);

        dim3 gg((N + 15) / 16, HC / 128);
        k_wmma_gemm<<<gg, TPB, 0, stream>>>(xin, Wp[l], nullptr, hbuf, nullptr,
                                            N, Kin[l], HC, 0);
        k_att<<<blocks((long)N * 32), TPB, 0, stream>>>(hbuf, asl[l], adl[l], a_s, a_d, N);
        k_edge_max<<<blocks((long)Etot * 4), TPB, 0, stream>>>(ei, a_s, a_d, ebuf, mmax, E, N);
        k_edge_exp<<<blocks((long)Etot * 4), TPB, 0, stream>>>(ei, ebuf, mmax, ssum, E, N);
        k_edge_scatter<<<blocks((long)Etot * 32), TPB, 0, stream>>>(ei, ebuf, ssum, hbuf, obuf, E, N);
        k_epilogue<<<blocks((long)N * HC), TPB, 0, stream>>>(obuf, bl[l], batch, xh, g, N, poolOff[l]);
    }

    // ---------------- pooled head: latent -> mu / log_var ------------------
    k_f2h<<<blocks((long)G * GCOLS), TPB, 0, stream>>>(g, g16, G * GCOLS);
    dim3 gl((G + 15) / 16, LATD / 128);
    k_wmma_gemm<<<gl, TPB, 0, stream>>>(g16, aggp, aggb, nullptr, lat16, G, GCOLS, LATD, 0);
    k_wmma_gemm<<<gl, TPB, 0, stream>>>(lat16, mup,  mub,  outp,              nullptr, G, LATD, LATD, 0);
    k_wmma_gemm<<<gl, TPB, 0, stream>>>(lat16, varp, varb, outp + (size_t)G * LATD, nullptr, G, LATD, LATD, 0);
}
